// BahdanauAttention_46531675685459
// MI455X (gfx1250) — matrix operand order repaired and checked by reading, not compiled, not run on hardware
//
#include <hip/hip_runtime.h>

#define B_ 32
#define T_ 4096
#define D_ 256
#define U_ 256

typedef __attribute__((ext_vector_type(16))) _Float16 v16h;
typedef __attribute__((ext_vector_type(8)))  float    v8f;
typedef __attribute__((ext_vector_type(4)))  unsigned u32x4;
typedef __attribute__((ext_vector_type(8)))  int      i32x8;
typedef __attribute__((ext_vector_type(4)))  int      i32x4;

// Native gfx1250 transcendental tanh (VOP1 v_tanh_f32).
__device__ __forceinline__ float fast_tanh(float x) {
    float r;
    asm("v_tanh_f32 %0, %1" : "=v"(r) : "v"(x));
    return r;
}

// ---------------------------------------------------------------------------
// Kernel 0: repack W1 (f32, [D,U] row-major) into f16 WMMA B-fragment layout.
// Fragment (nt, ks) covers B tile K=[ks*32, ks*32+32), N=[nt*16, nt*16+16).
// Contiguous-k order used here, matching A (16-bit B, 32x16): lane L holds column N = L%16;
// lanes 0-15 hold K=0..15, lanes 16-31 hold K=16..31 (16 halves per lane).
// ws layout: wsW1[((nt*8+ks)*32 + lane)*16 + e]
// ---------------------------------------------------------------------------
__global__ void pack_w1(const float* __restrict__ W1, _Float16* __restrict__ wsW1) {
    const int tile = blockIdx.x;          // 0..127  (nt*8 + ks)
    const int nt   = tile >> 3;
    const int ks   = tile & 7;
    const int lane = threadIdx.x;         // 0..31
    const int n    = lane & 15;
    const int kbase = ks * 32 + ((lane < 16) ? 0 : 16);
    _Float16* dst = wsW1 + ((size_t)tile * 32 + lane) * 16;
#pragma unroll
    for (int e = 0; e < 16; ++e)
        dst[e] = (_Float16)W1[(size_t)(kbase + e) * U_ + nt * 16 + n];
}

// ---------------------------------------------------------------------------
// Kernel 1: proj_hid[b,u] = hidden[b,:] . W2[:,u] + W2_b[u]     (tiny)
// ---------------------------------------------------------------------------
__global__ void proj_hidden(const float* __restrict__ hidden,
                            const float* __restrict__ W2,
                            const float* __restrict__ W2b,
                            float* __restrict__ ph) {
    const int b = blockIdx.x;
    const int u = threadIdx.x;
    float acc = W2b[u];
    const float* h = hidden + (size_t)b * D_;
#pragma unroll 4
    for (int d = 0; d < D_; ++d)
        acc = fmaf(h[d], W2[(size_t)d * U_ + u], acc);
    ph[(size_t)b * U_ + u] = acc;
}

// ---------------------------------------------------------------------------
// Kernel 2: WMMA GEMM + fused tanh/V epilogue.
// Packed W1 (128 KB) is DMA'd global->LDS once per workgroup by the Tensor
// Data Mover; inner-loop B fragments then come from LDS (ds_load_b128).
// Grid: (T/128, B), block 256 (8 waves). Wave w owns rows t0 = mt*128+w*16.
// ---------------------------------------------------------------------------
__global__ void __launch_bounds__(256)
scores_kernel(const float* __restrict__ enc,
              const _Float16* __restrict__ wsW1,
              const float* __restrict__ ph,
              const float* __restrict__ W1b,
              const float* __restrict__ Vw,
              const float* __restrict__ Vb,
              float* __restrict__ scores) {
    extern __shared__ _Float16 ldsW1[];    // 131072 bytes (dynamic LDS)

    const int b    = blockIdx.y;
    const int mt   = blockIdx.x;
    const int wave = threadIdx.x >> 5;
    const int lane = threadIdx.x & 31;
    const int hi   = (lane < 16) ? 0 : 1;  // which 16-lane half
    const int ln   = lane & 15;
    const int t0   = mt * 128 + wave * 16;

    // ---- TDM: one descriptor moves all 128 KB of packed W1 into LDS.
    // 1-row tile of 16384 8-byte elements (D# per ISA cdna5 8.3/8.4).
    if (wave == 0) {
        const unsigned lds_base = (unsigned)(size_t)(&ldsW1[0]);   // LDS byte addr
        const unsigned long long ga = (unsigned long long)(size_t)wsW1;

        u32x4 g0;
        g0[0] = 1u;                                   // count=1, load, user mode
        g0[1] = lds_base;                             // lds_addr [63:32]
        g0[2] = (unsigned)(ga & 0xFFFFFFFFull);       // global_addr low
        g0[3] = (unsigned)((ga >> 32) & 0x1FFFFFFull) // global_addr [120:96]
              | (2u << 30);                           // type=2 (image) [127:126]

        i32x8 g1;
        g1[0] = (int)(3u << 16);                      // data_size=3 (8B), mask=0
        g1[1] = (int)(16384u << 16);                  // tensor_dim0 lo16 @ [63:48]
        g1[2] = (int)(1u << 16);                      // tensor_dim0 hi=0; tensor_dim1=1
        g1[3] = (int)(16384u << 16);                  // tile_dim0 @ [127:112]
        g1[4] = 0;                                    // tile_dim1=0, tile_dim2=0
        g1[5] = (int)16384u;                          // tensor_dim0_stride lo32
        g1[6] = 0;                                    // stride hi / dim1_stride
        g1[7] = 0;

        i32x4 z4 = {0, 0, 0, 0};
        i32x8 z8 = {0, 0, 0, 0, 0, 0, 0, 0};
        __builtin_amdgcn_tensor_load_to_lds(g0, g1, z4, z4, z8, 0);
        __builtin_amdgcn_s_wait_tensorcnt((short)0);
    }
    __syncthreads();

    // A-matrix 16x32 f16 layout: lane holds row M = ln;
    // lanes 0-15: K = 0..15; lanes 16-31: K = 16..31
    const float* encRow = enc + ((size_t)b * T_ + (size_t)(t0 + ln)) * D_;
    const int klo = hi ? 16 : 0;

    // Per-lane V weights for the epilogue (C layout: lane column N = ln)
    float vwv[16];
#pragma unroll
    for (int nt = 0; nt < 16; ++nt) vwv[nt] = Vw[nt * 16 + ln];

    // Accumulators pre-loaded with bias = W1_b[u] + proj_hid[b,u]
    v8f acc[16];
#pragma unroll
    for (int nt = 0; nt < 16; ++nt) {
        const int u = nt * 16 + ln;
        const float bias = W1b[u] + ph[(size_t)b * U_ + u];
#pragma unroll
        for (int i = 0; i < 8; ++i) acc[nt][i] = bias;
    }

    for (int ks = 0; ks < 8; ++ks) {
        // ---- build A fragment: 4 coalesced b128 loads + f32->f16 converts
        const float4* p0 = (const float4*)(encRow + ks * 32 + klo);
        const float4  x0 = p0[0];
        const float4  x1 = p0[1];
        const float4* p1 = (const float4*)(encRow + ks * 32 + klo + 8);
        const float4  x2 = p1[0];
        const float4  x3 = p1[1];
        if (ks < 7)
            __builtin_prefetch(encRow + (ks + 1) * 32 + klo, 0, 0);

        v16h a;
        a[0]  = (_Float16)x0.x; a[1]  = (_Float16)x0.y;
        a[2]  = (_Float16)x0.z; a[3]  = (_Float16)x0.w;
        a[4]  = (_Float16)x1.x; a[5]  = (_Float16)x1.y;
        a[6]  = (_Float16)x1.z; a[7]  = (_Float16)x1.w;
        a[8]  = (_Float16)x2.x; a[9]  = (_Float16)x2.y;
        a[10] = (_Float16)x2.z; a[11] = (_Float16)x2.w;
        a[12] = (_Float16)x3.x; a[13] = (_Float16)x3.y;
        a[14] = (_Float16)x3.z; a[15] = (_Float16)x3.w;

        // ---- 16 WMMAs, B fragments from LDS (32B per lane -> ds_load_b128 x2)
#pragma unroll
        for (int nt = 0; nt < 16; ++nt) {
            const v16h bf =
                *(const v16h*)(ldsW1 + ((size_t)(nt * 8 + ks) * 32 + lane) * 16);
            acc[nt] = __builtin_amdgcn_wmma_f32_16x16x32_f16(
                false, a, false, bf, (short)0, acc[nt], false, false);
        }
    }

    // ---- epilogue: score[m] = sum_u tanh(acc) * Vw[u] + Vb
    // C layout: lane ln holds column N = nt*16+ln; VGPR r holds row M = r + hi*8.
    const float vb = Vb[0];
#pragma unroll
    for (int r = 0; r < 8; ++r) {
        float s = 0.0f;
#pragma unroll
        for (int nt = 0; nt < 16; ++nt)
            s = fmaf(fast_tanh(acc[nt][r]), vwv[nt], s);
        // reduce across the 16 lanes of this half (xor masks stay in-half)
        s += __shfl_xor(s, 1);
        s += __shfl_xor(s, 2);
        s += __shfl_xor(s, 4);
        s += __shfl_xor(s, 8);
        if (ln == 0)
            scores[(size_t)b * T_ + t0 + r + hi * 8] = s + vb;
    }
}

// ---------------------------------------------------------------------------
// Kernel 3: per-batch softmax over T=4096; writes attn weights to d_out.
// ---------------------------------------------------------------------------
__global__ void softmax_kernel(const float* __restrict__ scores,
                               float* __restrict__ attn) {
    const int b   = blockIdx.x;
    const int tid = threadIdx.x;       // 256 threads, 16 elems each
    __shared__ float red[256];
    const float* s = scores + (size_t)b * T_;

    float v[16];
    float m = -3.402823466e+38f;
#pragma unroll
    for (int i = 0; i < 16; ++i) {
        v[i] = s[tid + i * 256];
        m = fmaxf(m, v[i]);
    }
    red[tid] = m;
    __syncthreads();
    for (int off = 128; off > 0; off >>= 1) {
        if (tid < off) red[tid] = fmaxf(red[tid], red[tid + off]);
        __syncthreads();
    }
    m = red[0];
    __syncthreads();

    float sum = 0.0f;
#pragma unroll
    for (int i = 0; i < 16; ++i) {
        v[i] = __expf(v[i] - m);
        sum += v[i];
    }
    red[tid] = sum;
    __syncthreads();
    for (int off = 128; off > 0; off >>= 1) {
        if (tid < off) red[tid] += red[tid + off];
        __syncthreads();
    }
    const float inv = 1.0f / red[0];

#pragma unroll
    for (int i = 0; i < 16; ++i)
        attn[(size_t)b * T_ + tid + i * 256] = v[i] * inv;
}

// ---------------------------------------------------------------------------
// Kernel 4/5: context[b,d] = sum_t attn[b,t] * enc[b,t,d]  (enc re-read from L2)
// ---------------------------------------------------------------------------
__global__ void context_partial(const float* __restrict__ enc,
                                const float* __restrict__ attn,
                                float* __restrict__ part) {
    const int c = blockIdx.x;          // t-chunk 0..7 (512 rows each)
    const int b = blockIdx.y;
    const int d = threadIdx.x;         // 0..255
    const float* e = enc + ((size_t)b * T_ + (size_t)c * 512) * D_ + d;
    const float* w = attn + (size_t)b * T_ + (size_t)c * 512;
    float acc = 0.0f;
#pragma unroll 8
    for (int t = 0; t < 512; ++t)
        acc = fmaf(w[t], e[(size_t)t * D_], acc);
    part[((size_t)b * 8 + c) * D_ + d] = acc;
}

__global__ void context_final(const float* __restrict__ part,
                              float* __restrict__ ctx) {
    const int b = blockIdx.x;
    const int d = threadIdx.x;
    float acc = 0.0f;
#pragma unroll
    for (int c = 0; c < 8; ++c)
        acc += part[((size_t)b * 8 + c) * D_ + d];
    ctx[(size_t)b * D_ + d] = acc;
}

// ---------------------------------------------------------------------------
extern "C" void kernel_launch(void* const* d_in, const int* in_sizes, int n_in,
                              void* d_out, int out_size, void* d_ws, size_t ws_size,
                              hipStream_t stream) {
    (void)in_sizes; (void)n_in; (void)out_size; (void)ws_size;
    const float* enc    = (const float*)d_in[0];   // [32,4096,256]
    const float* hidden = (const float*)d_in[1];   // [32,256]
    const float* W1w    = (const float*)d_in[2];   // [256,256]
    const float* W1b    = (const float*)d_in[3];   // [256]
    const float* W2w    = (const float*)d_in[4];   // [256,256]
    const float* W2b    = (const float*)d_in[5];   // [256]
    const float* Vw     = (const float*)d_in[6];   // [256,1]
    const float* Vb     = (const float*)d_in[7];   // [1]

    float* out  = (float*)d_out;
    float* ctx  = out;                 // [32,256]
    float* attn = out + B_ * D_;       // [32,4096]

    char* ws = (char*)d_ws;
    _Float16* wsW1 = (_Float16*)ws;                                  // 131072 B
    float* ph      = (float*)(ws + 131072);                          //  32768 B
    float* scores  = (float*)(ws + 131072 + 32768);                  // 524288 B
    float* part    = (float*)(ws + 131072 + 32768 + 524288);         // 262144 B

    pack_w1<<<dim3(128), dim3(32), 0, stream>>>(W1w, wsW1);
    proj_hidden<<<dim3(B_), dim3(U_), 0, stream>>>(hidden, W2w, W2b, ph);
    scores_kernel<<<dim3(T_ / 128, B_), dim3(256), 131072, stream>>>(
        enc, wsW1, ph, W1b, Vw, Vb, scores);
    softmax_kernel<<<dim3(B_), dim3(256), 0, stream>>>(scores, attn);
    context_partial<<<dim3(8, B_), dim3(256), 0, stream>>>(enc, attn, part);
    context_final<<<dim3(B_), dim3(D_), 0, stream>>>(part, ctx);
}
